// SimpleMultiHeadAttention_14207751815619
// MI455X (gfx1250) — compile-verified, matrix operand order checked by reading, not run on hardware
//
#include <hip/hip_runtime.h>

// ---------------------------------------------------------------------------
// Multi-head attention for MI455X (gfx1250, wave32, WMMA f32_16x16x32_f16)
//   B=4, T=2048, D=768, H=12, hd=64
// Pipeline:
//   1) f32->f16 convert x; transpose+convert Wq/Wk/Wv/Wo
//   2) WMMA GEMM (32x64 tile/wave) -> Q,K in [B,H,T,64] f16 ;
//      V in [B,H,64,T] f16 (transposed)
//   3) flash-attention: 1 wave per 32-query tile (2 sub-tiles sharing K/V
//      fragments), keys streamed 32 at a time, online softmax, P relaid
//      C->A via per-wave LDS, PV via WMMA
//   4) WMMA GEMM attn_out @ Wo + bo -> f32 d_out
// ---------------------------------------------------------------------------

typedef _Float16 v16h __attribute__((ext_vector_type(16)));
typedef _Float16 v8h  __attribute__((ext_vector_type(8)));
typedef float    v8f  __attribute__((ext_vector_type(8)));

#define BB 4
#define TT 2048
#define DD 768
#define HH 12
#define HD 64
#define MM (BB * TT)           // 8192 rows

// GEMM wave-tile: 32 rows x 64 cols
#define WTM 32
#define WTN 64
#define GT_M (MM / WTM)        // 256
#define GT_N (DD / WTN)        // 12

// Workspace byte offsets
#define OFF_XH   0u
#define OFF_WQT  (OFF_XH  + (size_t)MM * DD * 2)
#define OFF_WKT  (OFF_WQT + (size_t)DD * DD * 2)
#define OFF_WVT  (OFF_WKT + (size_t)DD * DD * 2)
#define OFF_WOT  (OFF_WVT + (size_t)DD * DD * 2)
#define OFF_Q    (OFF_WOT + (size_t)DD * DD * 2)
#define OFF_K    (OFF_Q   + (size_t)MM * DD * 2)
#define OFF_VT   (OFF_K   + (size_t)MM * DD * 2)
#define OFF_AO   (OFF_VT  + (size_t)MM * DD * 2)

__device__ __forceinline__ v8f wmma16(v16h a, v16h b, v8f c) {
  // D = A(16x32,f16) x B(32x16,f16) + C(16x16,f32)
  return __builtin_amdgcn_wmma_f32_16x16x32_f16(
      /*neg_a=*/false, a, /*neg_b=*/false, b,
      /*c_mod=*/(short)0, c, /*reuse_a=*/false, /*reuse_b=*/false);
}

// Build a 16-half fragment from two contiguous 16-byte runs.
__device__ __forceinline__ v16h make_frag(const _Float16* p0, const _Float16* p1) {
  v8h lo = *(const v8h*)p0;
  v8h hi = *(const v8h*)p1;
  v16h r;
#pragma unroll
  for (int i = 0; i < 8; ++i) { r[i] = lo[i]; r[i + 8] = hi[i]; }
  return r;
}

// ---------------------------------------------------------------------------
__global__ void cvt_f32_to_f16_kernel(const float* __restrict__ in,
                                      _Float16* __restrict__ out, int n) {
  int i = blockIdx.x * blockDim.x + threadIdx.x;
  if (i < n) out[i] = (_Float16)in[i];
}

// Wt[n*768 + k] = (f16) W[k*768 + n]   (768x768)
__global__ void transpose_cvt_kernel(const float* __restrict__ in,
                                     _Float16* __restrict__ out) {
  int i = blockIdx.x * blockDim.x + threadIdx.x;
  if (i < DD * DD) {
    int k = i % DD;
    int n = i / DD;
    out[i] = (_Float16)in[k * DD + n];
  }
}

// ---------------------------------------------------------------------------
// WMMA GEMM, 32x64 tile per wave (2 M-subtiles x 4 N-subtiles):
//   A  : f16 row-major [M, 768]
//   Bt : f16 transposed weight [N=768, K=768] (row n = column n of W)
// mode 0: out f16 scatter [B,H,T,64]   (Q, K)
// mode 1: out f16 scatter [B,H,64,T]   (V transposed)
// mode 2: out f32 row-major [M, 768]   (final projection)
// ---------------------------------------------------------------------------
__global__ __launch_bounds__(256)
void gemm_wmma_kernel(const _Float16* __restrict__ A,
                      const _Float16* __restrict__ Bt,
                      const float* __restrict__ bias,
                      void* __restrict__ out, int mode) {
  const int tid  = threadIdx.x;
  const int lane = tid & 31;
  const int wave = tid >> 5;
  const int w    = blockIdx.x * 8 + wave;
  const int tm   = w % GT_M;
  const int tn   = w / GT_M;
  const int m0   = tm * WTM;
  const int n0   = tn * WTN;
  const int hl   = lane >> 4;   // half-wave id
  const int l16  = lane & 15;

  const _Float16* arow[2];
  const _Float16* brow[4];
#pragma unroll
  for (int mi = 0; mi < 2; ++mi)
    arow[mi] = A + (size_t)(m0 + mi * 16 + l16) * DD;
#pragma unroll
  for (int ni = 0; ni < 4; ++ni)
    brow[ni] = Bt + (size_t)(n0 + ni * 16 + l16) * DD;

  v8f acc[2][4];
#pragma unroll
  for (int mi = 0; mi < 2; ++mi)
#pragma unroll
    for (int ni = 0; ni < 4; ++ni)
      acc[mi][ni] = (v8f){0.f, 0.f, 0.f, 0.f, 0.f, 0.f, 0.f, 0.f};

  for (int k0 = 0; k0 < DD; k0 += 32) {
    __builtin_prefetch((const void*)(arow[0] + k0 + 256), 0, 1);
    __builtin_prefetch((const void*)(brow[0] + k0 + 256), 0, 1);
    v16h a0 = make_frag(arow[0] + k0 + hl * 8, arow[0] + k0 + 16 + hl * 8);
    v16h a1 = make_frag(arow[1] + k0 + hl * 8, arow[1] + k0 + 16 + hl * 8);
#pragma unroll
    for (int ni = 0; ni < 4; ++ni) {
      v16h b = make_frag(brow[ni] + k0 + hl * 8, brow[ni] + k0 + 16 + hl * 8);
      acc[0][ni] = wmma16(a0, b, acc[0][ni]);
      acc[1][ni] = wmma16(a1, b, acc[1][ni]);
    }
  }

#pragma unroll
  for (int ni = 0; ni < 4; ++ni) {
    const int n = n0 + ni * 16 + l16;     // output column (C/D: N = lane%16)
    const float bval = bias[n];
    const int h = n >> 6;                 // head (64-wide N-tile stays in head)
    const int d = n & 63;
#pragma unroll
    for (int mi = 0; mi < 2; ++mi) {
      if (mode == 2) {
        float* o = (float*)out;
#pragma unroll
        for (int r = 0; r < 8; ++r) {
          int m = m0 + mi * 16 + r + 8 * hl;   // C/D: M = r + 8*(lane/16)
          o[(size_t)m * DD + n] = acc[mi][ni][r] + bval;
        }
      } else {
        _Float16* o = (_Float16*)out;
#pragma unroll
        for (int r = 0; r < 8; ++r) {
          int m = m0 + mi * 16 + r + 8 * hl;
          int b = m >> 11;                     // T = 2048
          int t = m & 2047;
          size_t idx = (mode == 0)
                           ? ((size_t)((b * HH + h) * TT + t) * HD + d)
                           : ((size_t)((b * HH + h) * HD + d) * TT + t);
          o[idx] = (_Float16)(acc[mi][ni][r] + bval);
        }
      }
    }
  }
}

// ---------------------------------------------------------------------------
// Flash attention: one wave handles 32 query rows (2 sub-tiles) of one (b,h),
// sharing every K and V fragment between the two query sub-tiles.
//   Q,K : [B*H, T, 64] f16 ; Vt : [B*H, 64, T] f16 ; AO : [M, 768] f16
// ---------------------------------------------------------------------------
__global__ __launch_bounds__(256)
void flash_attn_kernel(const _Float16* __restrict__ Q,
                       const _Float16* __restrict__ K,
                       const _Float16* __restrict__ Vt,
                       _Float16* __restrict__ AO) {
  __shared__ _Float16 pbuf[8][2 * 16 * 32];   // per-wave P tiles (C->A relayout)

  const int tid  = threadIdx.x;
  const int lane = tid & 31;
  const int wave = tid >> 5;
  const int w    = blockIdx.x * 8 + wave;
  const int qt   = w & 63;                // T/32 = 64 query tiles
  const int bh   = w >> 6;                // 0..47
  const int hl   = lane >> 4;
  const int l16  = lane & 15;
  const int q0   = qt * 32;

  // Q A-fragments: [q-subtile][k-half of head dim]
  v16h qa[2][2];
#pragma unroll
  for (int q = 0; q < 2; ++q) {
    const _Float16* qrow = Q + ((size_t)bh * TT + q0 + q * 16 + l16) * HD;
    qa[q][0] = make_frag(qrow + hl * 8,      qrow + 16 + hl * 8);
    qa[q][1] = make_frag(qrow + 32 + hl * 8, qrow + 48 + hl * 8);
  }

  v8f o[2][4];
  float mrow[2][8], lrow[2][8];
#pragma unroll
  for (int q = 0; q < 2; ++q) {
#pragma unroll
    for (int ni = 0; ni < 4; ++ni)
      o[q][ni] = (v8f){0.f, 0.f, 0.f, 0.f, 0.f, 0.f, 0.f, 0.f};
#pragma unroll
    for (int r = 0; r < 8; ++r) { mrow[q][r] = -__builtin_inff(); lrow[q][r] = 0.f; }
  }

  _Float16* pb = pbuf[wave];
  const _Float16* kbase = K  + (size_t)bh * TT * HD;
  const _Float16* vbase = Vt + (size_t)bh * HD * TT;
  const float scale = 0.125f;             // 1/sqrt(64)

  for (int kt = 0; kt < TT; kt += 32) {
    // ---- S = Q @ K^T : 2 q-subtiles x 2 key-subtiles, K frags shared ----
    v8f s[2][2];
#pragma unroll
    for (int j = 0; j < 2; ++j) {
      const _Float16* kr = kbase + (size_t)(kt + j * 16 + l16) * HD;
      v16h kb0 = make_frag(kr + hl * 8,      kr + 16 + hl * 8);
      v16h kb1 = make_frag(kr + 32 + hl * 8, kr + 48 + hl * 8);
#pragma unroll
      for (int q = 0; q < 2; ++q) {
        v8f t = (v8f){0.f, 0.f, 0.f, 0.f, 0.f, 0.f, 0.f, 0.f};
        t = wmma16(qa[q][0], kb0, t);
        t = wmma16(qa[q][1], kb1, t);
        s[q][j] = t;
      }
    }

    // ---- online softmax over the 32 new keys, per q-subtile -------------
    float alpha[2][8];
#pragma unroll
    for (int q = 0; q < 2; ++q) {
#pragma unroll
      for (int r = 0; r < 8; ++r) {
        float s0 = s[q][0][r] * scale;
        float s1 = s[q][1][r] * scale;
        float mx = fmaxf(s0, s1);
        mx = fmaxf(mx, __shfl_xor(mx, 1, 32));     // reduce across 16 cols
        mx = fmaxf(mx, __shfl_xor(mx, 2, 32));     // (stays within half-wave)
        mx = fmaxf(mx, __shfl_xor(mx, 4, 32));
        mx = fmaxf(mx, __shfl_xor(mx, 8, 32));
        float mn = fmaxf(mrow[q][r], mx);
        alpha[q][r] = __expf(mrow[q][r] - mn);
        mrow[q][r] = mn;
        float p0 = __expf(s0 - mn);
        float p1 = __expf(s1 - mn);
        s[q][0][r] = p0;
        s[q][1][r] = p1;
        float rs = p0 + p1;
        rs += __shfl_xor(rs, 1, 32);
        rs += __shfl_xor(rs, 2, 32);
        rs += __shfl_xor(rs, 4, 32);
        rs += __shfl_xor(rs, 8, 32);
        lrow[q][r] = lrow[q][r] * alpha[q][r] + rs;
      }
      // relay P: C layout -> A layout via per-wave LDS
#pragma unroll
      for (int r = 0; r < 8; ++r) {
        int M = r + 8 * hl;
        pb[q * 512 + M * 32 + l16]      = (_Float16)s[q][0][r];
        pb[q * 512 + M * 32 + 16 + l16] = (_Float16)s[q][1][r];
      }
      // rescale O accumulators
#pragma unroll
      for (int ni = 0; ni < 4; ++ni)
#pragma unroll
        for (int r = 0; r < 8; ++r)
          o[q][ni][r] *= alpha[q][r];
    }

    v16h pa[2];
#pragma unroll
    for (int q = 0; q < 2; ++q) {
      const _Float16* prow = pb + q * 512 + l16 * 32;
      pa[q] = make_frag(prow + hl * 8, prow + 16 + hl * 8);  // 16x32 A-frag
    }

    // ---- O += P @ V : V frags shared between q-subtiles -----------------
#pragma unroll
    for (int ni = 0; ni < 4; ++ni) {
      const _Float16* vc = vbase + (size_t)(ni * 16 + l16) * TT + kt;
      v16h vb = make_frag(vc + hl * 8, vc + 16 + hl * 8);
      o[0][ni] = wmma16(pa[0], vb, o[0][ni]);
      o[1][ni] = wmma16(pa[1], vb, o[1][ni]);
    }
  }

  // ---- finalize: divide by l, store to [M, 768] f16 ----------------------
  const int b = bh / HH;
  const int h = bh % HH;
#pragma unroll
  for (int q = 0; q < 2; ++q) {
#pragma unroll
    for (int r = 0; r < 8; ++r) {
      float inv = 1.f / lrow[q][r];
      int M = r + 8 * hl;
      size_t base = (size_t)(b * TT + q0 + q * 16 + M) * DD + h * HD;
#pragma unroll
      for (int ni = 0; ni < 4; ++ni)
        AO[base + ni * 16 + l16] = (_Float16)(o[q][ni][r] * inv);
    }
  }
}

// ---------------------------------------------------------------------------
extern "C" void kernel_launch(void* const* d_in, const int* in_sizes, int n_in,
                              void* d_out, int out_size, void* d_ws, size_t ws_size,
                              hipStream_t stream) {
  const float* x  = (const float*)d_in[0];
  const float* Wq = (const float*)d_in[1];
  const float* bq = (const float*)d_in[2];
  const float* Wk = (const float*)d_in[3];
  const float* bk = (const float*)d_in[4];
  const float* Wv = (const float*)d_in[5];
  const float* bv = (const float*)d_in[6];
  const float* Wo = (const float*)d_in[7];
  const float* bo = (const float*)d_in[8];

  char* ws = (char*)d_ws;
  _Float16* xh  = (_Float16*)(ws + OFF_XH);
  _Float16* wqt = (_Float16*)(ws + OFF_WQT);
  _Float16* wkt = (_Float16*)(ws + OFF_WKT);
  _Float16* wvt = (_Float16*)(ws + OFF_WVT);
  _Float16* wot = (_Float16*)(ws + OFF_WOT);
  _Float16* qh  = (_Float16*)(ws + OFF_Q);
  _Float16* kh  = (_Float16*)(ws + OFF_K);
  _Float16* vt  = (_Float16*)(ws + OFF_VT);
  _Float16* ao  = (_Float16*)(ws + OFF_AO);

  const int nx = MM * DD;                              // 6,291,456
  cvt_f32_to_f16_kernel<<<(nx + 255) / 256, 256, 0, stream>>>(x, xh, nx);
  const int nw = DD * DD;                              // 589,824
  transpose_cvt_kernel<<<(nw + 255) / 256, 256, 0, stream>>>(Wq, wqt);
  transpose_cvt_kernel<<<(nw + 255) / 256, 256, 0, stream>>>(Wk, wkt);
  transpose_cvt_kernel<<<(nw + 255) / 256, 256, 0, stream>>>(Wv, wvt);
  transpose_cvt_kernel<<<(nw + 255) / 256, 256, 0, stream>>>(Wo, wot);

  const int gemm_blocks = (GT_M * GT_N) / 8;           // 384
  gemm_wmma_kernel<<<gemm_blocks, 256, 0, stream>>>(xh, wqt, bq, (void*)qh, 0);
  gemm_wmma_kernel<<<gemm_blocks, 256, 0, stream>>>(xh, wkt, bk, (void*)kh, 0);
  gemm_wmma_kernel<<<gemm_blocks, 256, 0, stream>>>(xh, wvt, bv, (void*)vt, 1);

  const int attn_blocks = (BB * HH * (TT / 32)) / 8;   // 384
  flash_attn_kernel<<<attn_blocks, 256, 0, stream>>>(qh, kh, vt, ao);

  gemm_wmma_kernel<<<gemm_blocks, 256, 0, stream>>>(ao, wot, bo, d_out, 2);
}
